// Decoder_14482629722137
// MI455X (gfx1250) — matrix operand order repaired and checked by reading, not compiled, not run on hardware
//
#include <hip/hip_runtime.h>
#include <hip/hip_bf16.h>

// ---------------------------------------------------------------------------
// CDNA5 (gfx1250, wave32) persistent double-LSTM decoder.
// All GEMMs use v_wmma_f32_16x16x32_bf16.
//
// Parallel decomposition: 8 batch groups (32 rows) x 4 column groups = 32
// persistent workgroups (512 thr = 16 waves each). h/emb/ts are exchanged
// through small L2-resident global buffers (double-buffered by step parity);
// the 4 column groups of a batch group sync with a global-atomic barrier.
// Cell state stays in registers for all 64 steps.
// ---------------------------------------------------------------------------

typedef __bf16 bf16;
typedef __attribute__((ext_vector_type(8)))  __bf16 v8bf;
typedef __attribute__((ext_vector_type(16))) __bf16 v16bf;
typedef __attribute__((ext_vector_type(8)))  float  v8f;

#define B_TOT 256
#define ZD    256
#define TD    512
#define AE    256
#define CFD   1024
#define NG    4096          // 4*CFD
#define NACT  64
#define STEPS 64
#define XA    768           // TD + AE
#define GROWS 32            // batch rows per group
#define NGRP  8             // batch groups
#define NCG   4             // column groups per batch group
#define NWG   (NGRP * NCG)  // 32 persistent workgroups

static __device__ __forceinline__ float sigmoidf_(float x) {
    return 1.0f / (1.0f + __expf(-x));
}
static __device__ __forceinline__ v16bf cat16(v8bf lo, v8bf hi) {
    return __builtin_shufflevector(lo, hi, 0,1,2,3,4,5,6,7,8,9,10,11,12,13,14,15);
}
// A-fragment (16x32 bf16): lane L = row M=L&15; lanes 0-15 hold K kb+0..15,
// lanes 16-31 hold kb+16..31.
static __device__ __forceinline__ v16bf load_A(const bf16* rowPtr, int kb, int sel) {
    v8bf lo = *(const v8bf*)(rowPtr + kb + sel * 16);
    v8bf hi = *(const v8bf*)(rowPtr + kb + 8 + sel * 16);
    return cat16(lo, hi);
}
// B-fragment (32x16 bf16): lane L = col N=L&15 (== weight row n); contiguous 16 K.
static __device__ __forceinline__ v16bf load_B(const bf16* rowPtr, int kb, int sel) {
    const bf16* p = rowPtr + kb + sel * 16;
    return cat16(*(const v8bf*)p, *(const v8bf*)(p + 8));
}
static __device__ __forceinline__ v8f wmma_bf16(v16bf a, v16bf b, v8f c) {
    return __builtin_amdgcn_wmma_f32_16x16x32_bf16(false, a, false, b, (short)0, c,
                                                   false, false);
}

// Global-atomic split barrier across the NCG workgroups of one batch group.
static __device__ __forceinline__ void group_barrier(unsigned int* bar,
                                                     unsigned int& rnd, int tid) {
    __syncthreads();
    rnd += 1;
    if (tid == 0) {
        __threadfence();                      // release h/emb/ts stores
        atomicAdd(bar, 1u);
        const unsigned int target = rnd * (unsigned int)NCG;
        while (__hip_atomic_load(bar, __ATOMIC_RELAXED, __HIP_MEMORY_SCOPE_AGENT)
               < target) {
            __builtin_amdgcn_s_sleep(1);
        }
        __threadfence();                      // acquire
    }
    __syncthreads();
}

// ---------------------------------------------------------------------------
// Prologue kernels
// ---------------------------------------------------------------------------

__global__ void k_trec(const float* __restrict__ z, const float* __restrict__ Wz,
                       const float* __restrict__ bz, bf16* __restrict__ trec) {
    int idx = blockIdx.x * blockDim.x + threadIdx.x;
    if (idx >= B_TOT * TD) return;
    int b = idx >> 9, j = idx & (TD - 1);
    const float* zr = z + b * ZD;
    const float* wr = Wz + j * ZD;
    float s = bz[j];
#pragma unroll 4
    for (int k = 0; k < ZD; ++k) s += zr[k] * wr[k];
    trec[idx] = (bf16)fmaxf(s, 0.0f);
}

__global__ void k_convert(const float* __restrict__ src, bf16* __restrict__ dst,
                          int rows, int srcStride, int srcOff, int cols, int dstStride) {
    int total = rows * cols;
    for (int i = blockIdx.x * blockDim.x + threadIdx.x; i < total;
         i += gridDim.x * blockDim.x) {
        int r = i / cols, c = i - r * cols;
        dst[(size_t)r * dstStride + c] = (bf16)src[(size_t)r * srcStride + srcOff + c];
    }
}

__global__ void k_col(const float* __restrict__ src, float* __restrict__ dst) {
    int n = blockIdx.x * blockDim.x + threadIdx.x;
    if (n < NG) dst[n] = src[(size_t)n * 769 + 768];
}

// pre = t_rec @ Wih[:, :512]^T + bih + bhh  (f32 [256 x 4096]), one wave per tile
__global__ void k_pregates(const bf16* __restrict__ trec, const bf16* __restrict__ Wih,
                           const float* __restrict__ bih, const float* __restrict__ bhh,
                           float* __restrict__ pre) {
    int wv = (blockIdx.x * blockDim.x + threadIdx.x) >> 5;  // 0..4095
    int lane = threadIdx.x & 31;
    int nTile = wv >> 4, mTile = wv & 15;
    int sel = lane >> 4, nl = lane & 15;
    int n = nTile * 16 + nl;
    const bf16* arow = trec + (size_t)(mTile * 16 + nl) * TD;
    const bf16* brow = Wih + (size_t)n * XA;
    float bias = bih[n] + bhh[n];
    v8f acc;
#pragma unroll
    for (int v = 0; v < 8; ++v) acc[v] = bias;
#pragma unroll 2
    for (int kb = 0; kb < TD; kb += 32)
        acc = wmma_bf16(load_A(arow, kb, sel), load_B(brow, kb, sel), acc);
#pragma unroll
    for (int v = 0; v < 8; ++v)
        pre[(size_t)(mTile * 16 + v + sel * 8) * NG + n] = acc[v];
}

// ---------------------------------------------------------------------------
// One LSTM gate phase for this wave: 16 h-cols (4 gate tiles) x 2 M-tiles.
// B-fragments are shared by both M-tiles (2 A-loads + 4 B-loads per 8 WMMAs).
// Reads h[old] / emb / ts from L2-resident buffers, writes h[new] (no race:
// old/new are distinct parity buffers).
// ---------------------------------------------------------------------------
static __device__ __forceinline__ void lstm_gates(
    const bf16* __restrict__ embG,   // [32 x 256]
    const bf16* __restrict__ hOld,   // [32 x 1024]
    bf16* __restrict__ hNew,         // [32 x 1024]
    const float* __restrict__ preG,  // [32 x NG]
    const bf16* __restrict__ Wih,    // [NG x 768], emb cols at +TD
    const bf16* __restrict__ Whh,    // [NG x 1024]
    const float* __restrict__ wcol,  // nullptr or ts column [NG]
    const float* __restrict__ tsPrev,// [32] (when wcol != nullptr)
    float* creg, int hcBase, int lane) {
    const int sel = lane >> 4, nl = lane & 15;
    v8f acc[4][2];
#pragma unroll
    for (int g = 0; g < 4; ++g) {
        const int n = g * CFD + hcBase + nl;
        const float w = wcol ? wcol[n] : 0.0f;
#pragma unroll
        for (int t = 0; t < 2; ++t) {
            v8f a;
#pragma unroll
            for (int v = 0; v < 8; ++v) {
                const int m = t * 16 + v + sel * 8;
                float x = preG[(size_t)m * NG + n];
                if (wcol) x += tsPrev[m] * w;
                a[v] = x;
            }
            acc[g][t] = a;
        }
    }
    {   // embedding contribution, K = 256
        const bf16* a0 = embG + (size_t)nl * AE;
        const bf16* a1 = embG + (size_t)(16 + nl) * AE;
#pragma unroll 2
        for (int kb = 0; kb < AE; kb += 32) {
            v16bf A0 = load_A(a0, kb, sel);
            v16bf A1 = load_A(a1, kb, sel);
#pragma unroll
            for (int g = 0; g < 4; ++g) {
                const int n = g * CFD + hcBase + nl;
                v16bf Bf = load_B(Wih + (size_t)n * XA + TD, kb, sel);
                acc[g][0] = wmma_bf16(A0, Bf, acc[g][0]);
                acc[g][1] = wmma_bf16(A1, Bf, acc[g][1]);
            }
        }
    }
    {   // hidden contribution, K = 1024
        const bf16* a0 = hOld + (size_t)nl * CFD;
        const bf16* a1 = hOld + (size_t)(16 + nl) * CFD;
#pragma unroll 2
        for (int kb = 0; kb < CFD; kb += 32) {
            v16bf A0 = load_A(a0, kb, sel);
            v16bf A1 = load_A(a1, kb, sel);
#pragma unroll
            for (int g = 0; g < 4; ++g) {
                const int n = g * CFD + hcBase + nl;
                v16bf Bf = load_B(Whh + (size_t)n * CFD, kb, sel);
                acc[g][0] = wmma_bf16(A0, Bf, acc[g][0]);
                acc[g][1] = wmma_bf16(A1, Bf, acc[g][1]);
            }
        }
    }
#pragma unroll
    for (int t = 0; t < 2; ++t)
#pragma unroll
        for (int v = 0; v < 8; ++v) {
            float iv = acc[0][t][v], fv = acc[1][t][v];
            float gv = acc[2][t][v], ov = acc[3][t][v];
            float c = sigmoidf_(fv) * creg[t * 8 + v] + sigmoidf_(iv) * tanhf(gv);
            creg[t * 8 + v] = c;
            float h = sigmoidf_(ov) * tanhf(c);
            hNew[(size_t)(t * 16 + v + sel * 8) * CFD + hcBase + nl] = (bf16)h;
        }
}

__global__ __launch_bounds__(512, 1) void k_decoder(
    const float* __restrict__ preA, const float* __restrict__ preT,
    const bf16* __restrict__ WihA, const bf16* __restrict__ WihT,
    const bf16* __restrict__ WhhA, const bf16* __restrict__ WhhT,
    const bf16* __restrict__ Wact, const float* __restrict__ b_act,
    const bf16* __restrict__ Wts1, const float* __restrict__ b_ts1,
    const float* __restrict__ wts2, const float* __restrict__ b_ts2,
    const bf16* __restrict__ Eb, const float* __restrict__ wcol,
    bf16* __restrict__ hbufA,   // [2][NGRP][32][1024]
    bf16* __restrict__ hbufT,   // [2][NGRP][32][1024]
    bf16* __restrict__ embbuf,  // [NGRP][32][256]
    float* __restrict__ tsbuf,  // [2][NGRP][32]
    unsigned int* __restrict__ barCnt,  // [NGRP]
    float* __restrict__ outActs, float* __restrict__ outTs) {
    __shared__ float s_logits[GROWS * NACT];
    __shared__ int   s_amax[GROWS];

    const int tid = threadIdx.x;
    const int wave = tid >> 5, lane = tid & 31;
    const int g  = blockIdx.x >> 2;   // batch group
    const int cg = blockIdx.x & 3;    // column group
    const int b0 = g * GROWS;
    const int hcBase = cg * 256 + wave * 16;
    const int sel = lane >> 4, nl = lane & 15;

    const float* preAg = preA + (size_t)b0 * NG;
    const float* preTg = preT + (size_t)b0 * NG;
    bf16* hA[2] = { hbufA + (size_t)g * GROWS * CFD,
                    hbufA + (size_t)(NGRP + g) * GROWS * CFD };
    bf16* hT[2] = { hbufT + (size_t)g * GROWS * CFD,
                    hbufT + (size_t)(NGRP + g) * GROWS * CFD };
    bf16*  embG   = embbuf + (size_t)g * GROWS * AE;
    float* tsb[2] = { tsbuf + g * GROWS, tsbuf + NGRP * GROWS + g * GROWS };
    unsigned int* bar = barCnt + g;
    unsigned int rnd = 0;

    // ---- init: zero this WG's h columns; leader inits emb (EOT) and ts ----
    for (int i = tid; i < GROWS * 256; i += 512) {
        const int r = i >> 8, c = cg * 256 + (i & 255);
        hA[0][(size_t)r * CFD + c] = (bf16)0.0f;
        hT[0][(size_t)r * CFD + c] = (bf16)0.0f;
    }
    if (cg == 0) {
        for (int i = tid; i < GROWS * AE; i += 512)
            embG[i] = Eb[(NACT - 1) * AE + (i & (AE - 1))];
        if (tid < GROWS) tsb[0][tid] = 0.0f;
    }
    float cA[16], cT[16];
#pragma unroll
    for (int i = 0; i < 16; ++i) { cA[i] = 0.0f; cT[i] = 0.0f; }
    group_barrier(bar, rnd, tid);

#pragma unroll 1
    for (int step = 0; step < STEPS; ++step) {
        const int p = step & 1, q = p ^ 1;
        // ---- activity LSTM: read hA[p]/emb, write hA[q] ----
        lstm_gates(embG, hA[p], hA[q], preAg, WihA, WhhA, nullptr, nullptr,
                   cA, hcBase, lane);
        group_barrier(bar, rnd, tid);                       // B1
        // ---- leader: logits, argmax, embedding gather, ts-acc init ----
        if (cg == 0) {
            if (wave < 8) {                    // 2 M-tiles x 4 col-tiles
                const int mt = wave >> 2, nt = wave & 3;
                const int n = nt * 16 + nl;
                v8f acc;
                const float bias = b_act[n];
#pragma unroll
                for (int v = 0; v < 8; ++v) acc[v] = bias;
                const bf16* arow = hA[q] + (size_t)(mt * 16 + nl) * CFD;
                const bf16* brow = Wact + (size_t)n * CFD;
#pragma unroll 2
                for (int kb = 0; kb < CFD; kb += 32)
                    acc = wmma_bf16(load_A(arow, kb, sel), load_B(brow, kb, sel), acc);
#pragma unroll
                for (int v = 0; v < 8; ++v)
                    s_logits[(mt * 16 + v + sel * 8) * NACT + n] = acc[v];
            }
            __syncthreads();
            if (tid < GROWS) {
                const float* lr = s_logits + tid * NACT;
                float best = lr[0]; int bi = 0;
                for (int a = 1; a < NACT; ++a) {
                    float v = lr[a];
                    if (v > best) { best = v; bi = a; }
                }
                s_amax[tid] = bi;
                tsb[q][tid] = b_ts2[0];        // init ts accumulation target
            }
            __syncthreads();
            for (int i = tid; i < GROWS * NACT; i += 512)
                outActs[((size_t)(b0 + (i >> 6)) * STEPS + step) * NACT + (i & 63)]
                    = s_logits[i];
            for (int i = tid; i < GROWS * AE; i += 512)
                embG[i] = Eb[(size_t)s_amax[i >> 8] * AE + (i & (AE - 1))];
        }
        group_barrier(bar, rnd, tid);                       // B2
        // ---- time LSTM: read hT[p]/emb(new)/ts[p], write hT[q] ----
        lstm_gates(embG, hT[p], hT[q], preTg, WihT, WhhT, wcol, tsb[p],
                   cT, hcBase, lane);
        group_barrier(bar, rnd, tid);                       // B3
        // ---- ts head: 64 tile-jobs (2 M x 32 col-tiles) over 4 WGs ----
        {
            const int j = cg * 16 + wave;      // 0..63
            const int mt = j >> 5, ct = j & 31;
            const int n = ct * 16 + nl;        // 0..511
            v8f acc;
            const float bias = b_ts1[n];
#pragma unroll
            for (int v = 0; v < 8; ++v) acc[v] = bias;
            const bf16* arow = hT[q] + (size_t)(mt * 16 + nl) * CFD;
            const bf16* brow = Wts1 + (size_t)n * CFD;
#pragma unroll 2
            for (int kb = 0; kb < CFD; kb += 32)
                acc = wmma_bf16(load_A(arow, kb, sel), load_B(brow, kb, sel), acc);
            const float w2 = wts2[n];
            float pr[8];
#pragma unroll
            for (int v = 0; v < 8; ++v) pr[v] = fmaxf(acc[v], 0.0f) * w2;
#pragma unroll
            for (int off = 1; off < 16; off <<= 1)
#pragma unroll
                for (int v = 0; v < 8; ++v) pr[v] += __shfl_xor(pr[v], off, 32);
            if (nl == 0)
#pragma unroll
                for (int v = 0; v < 8; ++v)
                    atomicAdd(&tsb[q][mt * 16 + v + sel * 8], pr[v]);
        }
        group_barrier(bar, rnd, tid);                       // B4
        if (cg == 0 && tid < GROWS)
            outTs[(size_t)(b0 + tid) * STEPS + step] = tsb[q][tid];
    }
}

// softmax over the activity axis, in place on d_out
__global__ void k_softmax(float* __restrict__ acts) {
    int idx = blockIdx.x * blockDim.x + threadIdx.x;
    if (idx >= B_TOT * STEPS) return;
    float* p = acts + (size_t)idx * NACT;
    float m = p[0];
    for (int a = 1; a < NACT; ++a) m = fmaxf(m, p[a]);
    float s = 0.0f;
    for (int a = 0; a < NACT; ++a) s += __expf(p[a] - m);
    float inv = 1.0f / s;
    for (int a = 0; a < NACT; ++a) p[a] = __expf(p[a] - m) * inv;
}

// ---------------------------------------------------------------------------

extern "C" void kernel_launch(void* const* d_in, const int* in_sizes, int n_in,
                              void* d_out, int out_size, void* d_ws, size_t ws_size,
                              hipStream_t stream) {
    (void)in_sizes; (void)n_in; (void)out_size; (void)ws_size;
    const float* z    = (const float*)d_in[0];
    const float* Wz2t = (const float*)d_in[1];
    const float* bz2t = (const float*)d_in[2];
    const float* E    = (const float*)d_in[3];
    const float* WihA = (const float*)d_in[4];
    const float* WhhA = (const float*)d_in[5];
    const float* bihA = (const float*)d_in[6];
    const float* bhhA = (const float*)d_in[7];
    const float* WihT = (const float*)d_in[8];
    const float* WhhT = (const float*)d_in[9];
    const float* bihT = (const float*)d_in[10];
    const float* bhhT = (const float*)d_in[11];
    const float* Wact = (const float*)d_in[12];
    const float* bact = (const float*)d_in[13];
    const float* Wts1 = (const float*)d_in[14];
    const float* bts1 = (const float*)d_in[15];
    const float* Wts2 = (const float*)d_in[16];
    const float* bts2 = (const float*)d_in[17];

    char* ws = (char*)d_ws;
    size_t off = 0;
    auto alloc = [&](size_t bytes) -> void* {
        void* p = ws + off;
        off += (bytes + 255) & ~(size_t)255;
        return p;
    };
    bf16*  trec  = (bf16*)alloc((size_t)B_TOT * TD * 2);
    bf16*  WihAb = (bf16*)alloc((size_t)NG * XA * 2);
    bf16*  WihTb = (bf16*)alloc((size_t)NG * XA * 2);
    bf16*  WhhAb = (bf16*)alloc((size_t)NG * CFD * 2);
    bf16*  WhhTb = (bf16*)alloc((size_t)NG * CFD * 2);
    bf16*  Wactb = (bf16*)alloc((size_t)NACT * CFD * 2);
    bf16*  Wts1b = (bf16*)alloc((size_t)(CFD / 2) * CFD * 2);
    bf16*  Ebuf  = (bf16*)alloc((size_t)NACT * AE * 2);
    float* wcol  = (float*)alloc((size_t)NG * 4);
    float* preA  = (float*)alloc((size_t)B_TOT * NG * 4);
    float* preT  = (float*)alloc((size_t)B_TOT * NG * 4);
    bf16*  hbufA = (bf16*)alloc((size_t)2 * NGRP * GROWS * CFD * 2);
    bf16*  hbufT = (bf16*)alloc((size_t)2 * NGRP * GROWS * CFD * 2);
    bf16*  embbf = (bf16*)alloc((size_t)NGRP * GROWS * AE * 2);
    float* tsbuf = (float*)alloc((size_t)2 * NGRP * GROWS * 4);
    unsigned int* barCnt = (unsigned int*)alloc(NGRP * 4);

    hipMemsetAsync(barCnt, 0, NGRP * 4, stream);

    k_trec<<<(B_TOT * TD + 255) / 256, 256, 0, stream>>>(z, Wz2t, bz2t, trec);
    k_convert<<<2048, 256, 0, stream>>>(WihA, WihAb, NG, XA, 0, XA, XA);
    k_convert<<<2048, 256, 0, stream>>>(WihT, WihTb, NG, XA + 1, 0, XA, XA);
    k_convert<<<2048, 256, 0, stream>>>(WhhA, WhhAb, NG, CFD, 0, CFD, CFD);
    k_convert<<<2048, 256, 0, stream>>>(WhhT, WhhTb, NG, CFD, 0, CFD, CFD);
    k_convert<<<256, 256, 0, stream>>>(Wact, Wactb, NACT, CFD, 0, CFD, CFD);
    k_convert<<<1024, 256, 0, stream>>>(Wts1, Wts1b, CFD / 2, CFD, 0, CFD, CFD);
    k_convert<<<64, 256, 0, stream>>>(E, Ebuf, NACT, AE, 0, AE, AE);
    k_col<<<(NG + 255) / 256, 256, 0, stream>>>(WihT, wcol);
    k_pregates<<<512, 256, 0, stream>>>(trec, WihAb, bihA, bhhA, preA);
    k_pregates<<<512, 256, 0, stream>>>(trec, WihTb, bihT, bhhT, preT);

    float* outActs = (float*)d_out;
    float* outTs   = outActs + (size_t)B_TOT * STEPS * NACT;
    k_decoder<<<NWG, 512, 0, stream>>>(preA, preT, WihAb, WihTb, WhhAb, WhhTb,
                                       Wactb, bact, Wts1b, bts1, Wts2, bts2,
                                       Ebuf, wcol, hbufA, hbufT, embbf, tsbuf,
                                       barCnt, outActs, outTs);
    k_softmax<<<(B_TOT * STEPS + 255) / 256, 256, 0, stream>>>(outActs);
}
